// SelfAttention_45930380264046
// MI455X (gfx1250) — matrix operand order repaired and checked by reading, not compiled, not run on hardware
//
#include <hip/hip_runtime.h>
#include <hip/hip_bf16.h>

#define EMBED 1024
#define HEADS 16
#define HDIM  64
#define NBATCH 4
#define SEQLEN 2048

typedef __bf16 bf16_t;
typedef bf16_t v16bf __attribute__((ext_vector_type(16)));
typedef bf16_t v8bf  __attribute__((ext_vector_type(8)));
typedef float  v8f   __attribute__((ext_vector_type(8)));
typedef unsigned int u32x4 __attribute__((ext_vector_type(4)));
typedef int i32x4 __attribute__((ext_vector_type(4)));
typedef int i32x8 __attribute__((ext_vector_type(8)));

#if __has_builtin(__builtin_amdgcn_tensor_load_to_lds)
#define HAVE_TDM 1
#else
#define HAVE_TDM 0
#endif

#if __has_builtin(__builtin_amdgcn_s_wait_tensorcnt)
#define WAIT_TENSOR(n) __builtin_amdgcn_s_wait_tensorcnt(n)
#else
#define WAIT_TENSOR(n) asm volatile("s_wait_tensorcnt %0" ::"n"(n) : "memory")
#endif

// ---- fragment helpers -------------------------------------------------------
// A-matrix 16x32 bf16 (ISA 7.12.2): lane m=lane%16, half=lane>>4,
//   element e -> K = 16*(e>>3) + 8*half + (e&7)  => two contiguous 8-half runs.
// B-matrix 32x16 bf16: lane n=lane%16, khalf=lane>>4, element e -> K = 16*(e>>3) + 8*khalf + (e&7)
//   => two contiguous 8-half runs, same order as A (one v8bf load each).
// C/D 16x16 f32: element r -> M = r + 8*(lane>>4), N = lane%16.

__device__ __forceinline__ v16bf frag_cat(v8bf lo, v8bf hi) {
  return __builtin_shufflevector(lo, hi, 0,1,2,3,4,5,6,7,8,9,10,11,12,13,14,15);
}
__device__ __forceinline__ v16bf frag_from_bf16(const bf16_t* p0, const bf16_t* p1) {
  return frag_cat(*(const v8bf*)p0, *(const v8bf*)p1);
}
__device__ __forceinline__ v8bf cvt8(const float* p) {
  v8bf r;
#pragma unroll
  for (int i = 0; i < 8; ++i) r[i] = (bf16_t)p[i];
  return r;
}
__device__ __forceinline__ v16bf frag_from_f32(const float* p0, const float* p1) {
  return frag_cat(cvt8(p0), cvt8(p1));
}
__device__ __forceinline__ v8f wmma_bf16(v16bf a, v16bf b, v8f c) {
  return __builtin_amdgcn_wmma_f32_16x16x32_bf16(false, a, false, b, (short)0, c,
                                                 false, false);
}

// ---- TDM: DMA one 64x64 bf16 tile (row-major, contiguous rows of 64) --------
// D# per ISA 8.3/8.4: group0 = {count=1 | lds_addr | global_addr | type=2},
// group1 = {wg_mask=0, data_size=2B, tensor_dim0=64, tensor_dim1=SEQLEN,
//           tile_dim0=64, tile_dim1=64, tensor_dim0_stride=64}.
__device__ __forceinline__ void tdm_load_tile_64x64(const bf16_t* gsrc, bf16_t* ldst) {
#if HAVE_TDM
  const unsigned long long ga = (unsigned long long)(uintptr_t)gsrc;
  const unsigned int la = (unsigned int)(uintptr_t)ldst;  // LDS addr = addr[31:0]
  u32x4 g0 = {1u, la, (unsigned int)ga,
              (unsigned int)((ga >> 32) & 0x1FFFFFFu) | 0x80000000u};
  i32x8 g1 = {0x00010000, 0x00400000, 0x08000000, 0x00400000, 64, 64, 0, 0};
  i32x4 gz = {0, 0, 0, 0};
#if defined(__clang_major__) && __clang_major__ >= 23
  i32x8 gz8 = {0, 0, 0, 0, 0, 0, 0, 0};
  __builtin_amdgcn_tensor_load_to_lds(g0, g1, gz, gz, gz8, 0);
#else
  __builtin_amdgcn_tensor_load_to_lds(g0, g1, gz, gz, 0);
#endif
#else
  (void)gsrc; (void)ldst;
#endif
}

// fallback cooperative copy, all 128 threads
__device__ __forceinline__ void copy_tile_64x64(const bf16_t* g, bf16_t* l, int tid) {
#pragma unroll
  for (int i = 0; i < 4; ++i)
    *((v8bf*)l + tid + i * 128) = *((const v8bf*)g + tid + i * 128);
}

// ---- kernel 1: per-head input projections (x @ W^T), f32 -> bf16 ------------
__global__ __launch_bounds__(128)
void proj_kernel(const float* __restrict__ xq, const float* __restrict__ xk,
                 const float* __restrict__ xv,
                 const float* __restrict__ wq, const float* __restrict__ wk,
                 const float* __restrict__ wv,
                 bf16_t* __restrict__ oq, bf16_t* __restrict__ ok,
                 bf16_t* __restrict__ ov)
{
  const int lane = threadIdx.x & 31;
  const int wave = threadIdx.x >> 5;
  const int h = blockIdx.y;
  const int which = blockIdx.z;

  const float* x = (which == 0) ? xq : (which == 1) ? xk : xv;
  const float* w = (which == 0) ? wq : (which == 1) ? wk : wv;
  bf16_t*      o = (which == 0) ? oq : (which == 1) ? ok : ov;

  const int m  = lane & 15;
  const int hf = lane >> 4;
  const int t0 = blockIdx.x * 16;

  const float* arow = x + (size_t)(t0 + m) * EMBED + h * HDIM;
  const float* brow = w + (size_t)(wave * 16 + m) * HDIM;

  v8f c = {0.f, 0.f, 0.f, 0.f, 0.f, 0.f, 0.f, 0.f};
#pragma unroll
  for (int kk = 0; kk < HDIM; kk += 32) {
    v16bf a = frag_from_f32(arow + kk + 8 * hf, arow + kk + 16 + 8 * hf);
    v16bf b = frag_from_f32(brow + kk + 8 * hf, brow + kk + 16 + 8 * hf);
    c = wmma_bf16(a, b, c);
  }

  const int n  = t0 / SEQLEN;
  const int l0 = t0 % SEQLEN;
  bf16_t* obase = o + ((size_t)n * HEADS + h) * SEQLEN * HDIM;
#pragma unroll
  for (int r = 0; r < 8; ++r)
    obase[(size_t)(l0 + r + 8 * hf) * HDIM + (wave * 16 + m)] = (bf16_t)c[r];
}

// ---- kernel 2: flash attention, 64 queries / workgroup, 64-key blocks -------
// grid: (SEQ/64, HEADS, N), block 128. Wave w owns query rows q0..q0+15.
// K/V tiles double-buffered in LDS via TDM (tensor_load_to_lds + TENSORcnt).
__global__ __launch_bounds__(128)
void attn_kernel(const bf16_t* __restrict__ qp, const bf16_t* __restrict__ kp,
                 const bf16_t* __restrict__ vp, const int* __restrict__ mask,
                 bf16_t* __restrict__ oa)
{
  __shared__ __align__(16) bf16_t kt[2][64][64];    // K tiles, row-major
  __shared__ __align__(16) bf16_t vraw[2][64][64];  // V tiles, row-major
  __shared__ __align__(16) bf16_t vt[64][72];       // V^T: [d][key], padded
  __shared__ __align__(16) bf16_t pt[4][16][72];    // per-wave P: [q][key]

  const int lane = threadIdx.x & 31;
  const int wave = threadIdx.x >> 5;
  const int h = blockIdx.y;
  const int n = blockIdx.z;
  const int m  = lane & 15;
  const int hf = lane >> 4;

  const size_t headoff = ((size_t)n * HEADS + h) * SEQLEN * HDIM;
  const bf16_t* Q = qp + headoff;
  const bf16_t* K = kp + headoff;
  const bf16_t* V = vp + headoff;
  const int* Mk = mask + (size_t)n * SEQLEN * SEQLEN;

  const int q0 = blockIdx.x * 64 + wave * 16;

  // Q fragments resident in registers for the whole key loop.
  const bf16_t* qrow = Q + (size_t)(q0 + m) * HDIM;
  v16bf qa0 = frag_from_bf16(qrow + 8 * hf,      qrow + 16 + 8 * hf);
  v16bf qa1 = frag_from_bf16(qrow + 32 + 8 * hf, qrow + 48 + 8 * hf);

  float mi[8], li[8], alpha[8];
  v8f acc[4];
#pragma unroll
  for (int r = 0; r < 8; ++r) { mi[r] = -__builtin_inff(); li[r] = 0.f; }
#pragma unroll
  for (int j = 0; j < 4; ++j) {
    v8f z = {0.f, 0.f, 0.f, 0.f, 0.f, 0.f, 0.f, 0.f};
    acc[j] = z;
  }

  const float scale = 0.03125f;  // 1/sqrt(EMBED)

  // prologue: stage block 0 into buffer 0
#if HAVE_TDM
  if (wave == 0) {
    tdm_load_tile_64x64(K, &kt[0][0][0]);
    tdm_load_tile_64x64(V, &vraw[0][0][0]);
  }
#else
  copy_tile_64x64(K, &kt[0][0][0], threadIdx.x);
  copy_tile_64x64(V, &vraw[0][0][0], threadIdx.x);
#endif

  for (int kb = 0, it = 0; kb < SEQLEN; kb += 64, ++it) {
    const int cur = it & 1;
    __syncthreads();  // prior block's vt/kt/vraw reads are done

    // issue DMA for the next block into the other buffer, then wait for cur
    if (kb + 64 < SEQLEN) {
#if HAVE_TDM
      if (wave == 0) {
        tdm_load_tile_64x64(K + (size_t)(kb + 64) * HDIM, &kt[cur ^ 1][0][0]);
        tdm_load_tile_64x64(V + (size_t)(kb + 64) * HDIM, &vraw[cur ^ 1][0][0]);
        WAIT_TENSOR(2);  // 4 outstanding, in-order -> oldest 2 (cur) complete
      }
#else
      copy_tile_64x64(K + (size_t)(kb + 64) * HDIM, &kt[cur ^ 1][0][0], threadIdx.x);
      copy_tile_64x64(V + (size_t)(kb + 64) * HDIM, &vraw[cur ^ 1][0][0], threadIdx.x);
#endif
    } else {
#if HAVE_TDM
      if (wave == 0) WAIT_TENSOR(0);
#endif
    }
    __syncthreads();  // cur K/V tiles visible to all waves

    // cooperative transpose: vraw[cur][key][d] -> vt[d][key]
    {
      const int key = threadIdx.x & 63;
      const int d0  = (threadIdx.x >> 6) * 32;
      const bf16_t* vr = &vraw[cur][key][d0];
#pragma unroll
      for (int i = 0; i < 32; i += 8) {
        v8bf a = *(const v8bf*)(vr + i);
#pragma unroll
        for (int u = 0; u < 8; ++u) vt[d0 + i + u][key] = a[u];
      }
    }
    __syncthreads();  // vt ready

    // S = Q K^T : four 16-key sub-tiles, K-dim 64 in two wmma steps each
    v8f s[4];
#pragma unroll
    for (int j = 0; j < 4; ++j) {
      const bf16_t* krow = &kt[cur][j * 16 + m][0];
      v16bf b0 = frag_from_bf16(krow + 8 * hf,      krow + 16 + 8 * hf);
      v16bf b1 = frag_from_bf16(krow + 32 + 8 * hf, krow + 48 + 8 * hf);
      v8f z = {0.f, 0.f, 0.f, 0.f, 0.f, 0.f, 0.f, 0.f};
      z = wmma_bf16(qa0, b0, z);
      s[j] = wmma_bf16(qa1, b1, z);
    }

    // scale + mask
#pragma unroll
    for (int j = 0; j < 4; ++j) {
      const int kcol = kb + j * 16 + m;
#pragma unroll
      for (int r = 0; r < 8; ++r) {
        const int qidx = q0 + r + 8 * hf;
        float v = s[j][r] * scale;
        if (Mk[(size_t)qidx * SEQLEN + kcol] == 0) v = -1.0e20f;
        s[j][r] = v;
      }
    }

    // online softmax across 64 keys (row stats across 16 lanes per half-wave)
#pragma unroll
    for (int r = 0; r < 8; ++r) {
      float rm = fmaxf(fmaxf(s[0][r], s[1][r]), fmaxf(s[2][r], s[3][r]));
#pragma unroll
      for (int sh = 1; sh < 16; sh <<= 1)
        rm = fmaxf(rm, __shfl_xor(rm, sh, 32));
      const float mnew = fmaxf(mi[r], rm);
      alpha[r] = __expf(mi[r] - mnew);
      mi[r] = mnew;
      float rs = 0.f;
#pragma unroll
      for (int j = 0; j < 4; ++j) {
        const float p = __expf(s[j][r] - mnew);
        s[j][r] = p;
        rs += p;
      }
#pragma unroll
      for (int sh = 1; sh < 16; sh <<= 1)
        rs += __shfl_xor(rs, sh, 32);
      li[r] = li[r] * alpha[r] + rs;
    }

    // stage P (C layout -> row-major [q][key]) for re-read as A-fragments
#pragma unroll
    for (int j = 0; j < 4; ++j)
#pragma unroll
      for (int r = 0; r < 8; ++r)
        pt[wave][r + 8 * hf][j * 16 + m] = (bf16_t)s[j][r];

#pragma unroll
    for (int j = 0; j < 4; ++j)
#pragma unroll
      for (int r = 0; r < 8; ++r)
        acc[j][r] *= alpha[r];

    const bf16_t* prow = &pt[wave][m][0];
    v16bf pa0 = frag_from_bf16(prow + 8 * hf,      prow + 16 + 8 * hf);
    v16bf pa1 = frag_from_bf16(prow + 32 + 8 * hf, prow + 48 + 8 * hf);

    // O += P @ V  (4 d-tiles of 16, keys 0..31 then 32..63)
#pragma unroll
    for (int j = 0; j < 4; ++j) {
      const bf16_t* vrow = &vt[j * 16 + m][0];
      v16bf vb0 = frag_from_bf16(vrow + 8 * hf,      vrow + 16 + 8 * hf);
      v16bf vb1 = frag_from_bf16(vrow + 32 + 8 * hf, vrow + 48 + 8 * hf);
      acc[j] = wmma_bf16(pa0, vb0, acc[j]);
      acc[j] = wmma_bf16(pa1, vb1, acc[j]);
    }
  }

  // normalize and store to [n][q][h*64+d] bf16 workspace
#pragma unroll
  for (int j = 0; j < 4; ++j)
#pragma unroll
    for (int r = 0; r < 8; ++r) {
      const float v = acc[j][r] / li[r];
      const int qidx = q0 + r + 8 * hf;
      const int col  = h * HDIM + j * 16 + m;
      oa[((size_t)n * SEQLEN + qidx) * EMBED + col] = (bf16_t)v;
    }
}

// ---- kernel 3: out = X @ Wo^T + bo, f32 result ------------------------------
__global__ __launch_bounds__(128)
void oproj_kernel(const bf16_t* __restrict__ x, const float* __restrict__ wo,
                  const float* __restrict__ bo, float* __restrict__ out)
{
  const int lane = threadIdx.x & 31;
  const int wave = threadIdx.x >> 5;
  const int m  = lane & 15;
  const int hf = lane >> 4;

  const int t0 = blockIdx.x * 16;
  const int c0 = blockIdx.y * 64 + wave * 16;

  const bf16_t* arow = x + (size_t)(t0 + m) * EMBED;
  const float*  brow = wo + (size_t)(c0 + m) * EMBED;

  v8f c = {0.f, 0.f, 0.f, 0.f, 0.f, 0.f, 0.f, 0.f};
  for (int kk = 0; kk < EMBED; kk += 32) {
    v16bf a = frag_from_bf16(arow + kk + 8 * hf, arow + kk + 16 + 8 * hf);
    v16bf b = frag_from_f32(brow + kk + 8 * hf, brow + kk + 16 + 8 * hf);
    c = wmma_bf16(a, b, c);
  }

  const float bias = bo[c0 + m];
#pragma unroll
  for (int r = 0; r < 8; ++r)
    out[(size_t)(t0 + r + 8 * hf) * EMBED + (c0 + m)] = c[r] + bias;
}

// ---- launch -----------------------------------------------------------------
extern "C" void kernel_launch(void* const* d_in, const int* in_sizes, int n_in,
                              void* d_out, int out_size, void* d_ws, size_t ws_size,
                              hipStream_t stream) {
  (void)in_sizes; (void)n_in; (void)out_size;
  const float* values = (const float*)d_in[0];
  const float* keys   = (const float*)d_in[1];
  const float* query  = (const float*)d_in[2];
  const int*   mask   = (const int*)d_in[3];
  const float* Wv     = (const float*)d_in[4];
  const float* Wk     = (const float*)d_in[5];
  const float* Wq     = (const float*)d_in[6];
  const float* Wo     = (const float*)d_in[7];
  const float* bo     = (const float*)d_in[8];
  float* out = (float*)d_out;

  const size_t per = (size_t)NBATCH * HEADS * SEQLEN * HDIM;
  const size_t need = (per * 3 + (size_t)NBATCH * SEQLEN * EMBED) * sizeof(bf16_t);
  if (ws_size < need) return;

  bf16_t* wsq = (bf16_t*)d_ws;
  bf16_t* wsk = wsq + per;
  bf16_t* wsv = wsk + per;
  bf16_t* wsa = wsv + per;

  const int ntok_tiles = NBATCH * SEQLEN / 16;  // 512

  dim3 g1(ntok_tiles, HEADS, 3);
  proj_kernel<<<g1, 128, 0, stream>>>(query, keys, values, Wq, Wk, Wv,
                                      wsq, wsk, wsv);

  dim3 g2(SEQLEN / 64, HEADS, NBATCH);
  attn_kernel<<<g2, 128, 0, stream>>>(wsq, wsk, wsv, mask, wsa);

  dim3 g3(ntok_tiles, EMBED / 64);
  oproj_kernel<<<g3, 128, 0, stream>>>(wsa, Wo, bo, out);
}
